// MeowBus_26130581029056
// MI455X (gfx1250) — matrix operand order repaired and checked by reading, not compiled, not run on hardware
//
#include <hip/hip_runtime.h>
#include <math.h>

// ---------------------------------------------------------------------------
// VQ-VAE forward for MI455X (gfx1250, wave32, WMMA).
// encoded = h @ enc_w^T + enc_b          (bf16 WMMA, f32 accum)
// dist    = e_sq - 2 * flat @ cb^T       (bf16 WMMA cross term) -> argmin
// message = quantized @ dec_w^T + dec_b  (bf16 WMMA, A gathered from codebook)
// ---------------------------------------------------------------------------

typedef __bf16 bf16_t;
typedef bf16_t v16bf __attribute__((ext_vector_type(16)));
typedef bf16_t v8bf  __attribute__((ext_vector_type(8)));
typedef float  v8f   __attribute__((ext_vector_type(8)));

#define WMMA_BF16(a, b, c) \
  __builtin_amdgcn_wmma_f32_16x16x32_bf16(false, (a), false, (b), (short)0, (c), false, false)

// A-matrix fragment (16x32 bf16), layout per CDNA5 ISA 7.12.2:
// lane m=lane&15, kh=lane>>4; elem i<8: K=kh*8+i ; elem i>=8: K=16+kh*8+(i-8)
__device__ __forceinline__ v16bf load_fragA(const bf16_t* base, int stride) {
  const int lane = threadIdx.x & 31;
  const int m    = lane & 15;
  const int kh   = lane >> 4;
  const bf16_t* row = base + m * stride + kh * 8;
  v8bf lo = *(const v8bf*)(row);
  v8bf hi = *(const v8bf*)(row + 16);
  v16bf r;
#pragma unroll
  for (int i = 0; i < 8; ++i) { r[i] = lo[i]; r[i + 8] = hi[i]; }
  return r;
}

// B-matrix fragment (32x16 bf16): n=lane&15, kh=lane>>4; elem i<8: K=kh*8+i ; elem i>=8: K=16+kh*8+(i-8)
__device__ __forceinline__ v16bf load_fragB(const bf16_t* base, int stride) {
  const int lane = threadIdx.x & 31;
  const int n    = lane & 15;
  const int kh   = lane >> 4;
  const bf16_t* row = base + n * stride + kh * 8;
  v8bf lo = *(const v8bf*)(row);
  v8bf hi = *(const v8bf*)(row + 16);
  v16bf r;
#pragma unroll
  for (int i = 0; i < 8; ++i) { r[i] = lo[i]; r[i + 8] = hi[i]; }
  return r;
}

// ---------------------------------------------------------------------------
// Tiled GEMM: C[M x N] = A[M x K] * W[N x K]^T + bias[N]
// GATHER=true: A row m is concat of codebook[gidx[m*4+s]][0:128], s=0..3
// Block tile 128x128, BK=32, 256 threads = 8 waves (4 M x 2 N), wave = 32x64.
// ---------------------------------------------------------------------------
template <bool GATHER>
__global__ void __launch_bounds__(256)
gemm_bf16_kernel(const float* __restrict__ A, const int* __restrict__ gidx,
                 const float* __restrict__ cb, const float* __restrict__ W,
                 const float* __restrict__ bias, float* __restrict__ C,
                 int M, int N, int K) {
  constexpr int SA = 40;  // padded LDS row stride (32 used + 8 pad), 80B rows
  __shared__ bf16_t As[128 * SA];
  __shared__ bf16_t Bs[128 * SA];

  const int t    = threadIdx.x;
  const int lane = t & 31;
  const int w    = t >> 5;
  const int wm   = w >> 1;   // 0..3
  const int wn   = w & 1;    // 0..1
  const int m0   = blockIdx.y * 128;
  const int n0   = blockIdx.x * 128;

  v8f acc[2][4];
#pragma unroll
  for (int i = 0; i < 2; ++i)
#pragma unroll
    for (int j = 0; j < 4; ++j)
#pragma unroll
      for (int r = 0; r < 8; ++r) acc[i][j][r] = 0.0f;

  for (int kb = 0; kb < K; kb += 32) {
    // ---- stage A tile: 128 rows x 32 cols (fp32 -> bf16) ----
#pragma unroll
    for (int i = 0; i < 4; ++i) {
      int q   = t + i * 256;       // float4 index, 1024 total
      int row = q >> 3;            // 8 float4 per row
      int c4  = (q & 7) << 2;
      float4 v;
      if (GATHER) {
        int seg = (kb + c4) >> 7;
        int kc  = (kb + c4) & 127;
        int id  = gidx[(m0 + row) * 4 + seg];
        v = *(const float4*)&cb[(size_t)id * 128 + kc];
      } else {
        v = *(const float4*)&A[(size_t)(m0 + row) * K + kb + c4];
      }
      bf16_t* d = &As[row * SA + c4];
      d[0] = (bf16_t)v.x; d[1] = (bf16_t)v.y; d[2] = (bf16_t)v.z; d[3] = (bf16_t)v.w;
    }
    // ---- stage B tile: Bs[n][k] = W[n0+n][kb+k] ----
#pragma unroll
    for (int i = 0; i < 4; ++i) {
      int q   = t + i * 256;
      int row = q >> 3;
      int c4  = (q & 7) << 2;
      float4 v = *(const float4*)&W[(size_t)(n0 + row) * K + kb + c4];
      bf16_t* d = &Bs[row * SA + c4];
      d[0] = (bf16_t)v.x; d[1] = (bf16_t)v.y; d[2] = (bf16_t)v.z; d[3] = (bf16_t)v.w;
    }
    __syncthreads();

    v16bf a0 = load_fragA(&As[(wm * 32) * SA], SA);
    v16bf a1 = load_fragA(&As[(wm * 32 + 16) * SA], SA);
#pragma unroll
    for (int j = 0; j < 4; ++j) {
      v16bf b = load_fragB(&Bs[(wn * 64 + j * 16) * SA], SA);
      acc[0][j] = WMMA_BF16(a0, b, acc[0][j]);
      acc[1][j] = WMMA_BF16(a1, b, acc[1][j]);
    }
    __syncthreads();
  }

  // ---- epilogue: C/D layout: VGPR r -> row r + 8*(lane>>4), col lane&15 ----
  const int lc = lane & 15;
  const int lh = lane >> 4;
#pragma unroll
  for (int i = 0; i < 2; ++i) {
#pragma unroll
    for (int j = 0; j < 4; ++j) {
      int gn = n0 + wn * 64 + j * 16 + lc;
      float bz = bias[gn];
      int gm = m0 + wm * 32 + i * 16 + lh * 8;
#pragma unroll
      for (int r = 0; r < 8; ++r) {
        C[(size_t)(gm + r) * N + gn] = acc[i][j][r] + bz;
      }
    }
  }
}

// ---------------------------------------------------------------------------
// VQ kernel: 128 flat rows per block; wave w owns rows [w*16, w*16+16).
// dist(row, code) = e_sq[code] - 2 * <flat[row], cb[code]>  (x^2 const/row)
// ---------------------------------------------------------------------------
__global__ void __launch_bounds__(256)
vq_kernel(const float* __restrict__ enc,   // [131072 x 128] fp32
          const float* __restrict__ cb,    // [512 x 128]
          const float* __restrict__ esq,   // [512]
          int* __restrict__ idx_out,       // [131072] ws
          float* __restrict__ idx_outf,    // [131072] d_out (float)
          float* __restrict__ counts,      // [512]
          float* __restrict__ mse_acc) {
  constexpr int SA = 136;  // 128 + 8 pad
  __shared__ bf16_t As[128 * SA];
  __shared__ bf16_t Bs[64 * SA];
  __shared__ int idx_sh[128];

  const int t    = threadIdx.x;
  const int lane = t & 31;
  const int w    = t >> 5;
  const int row0 = blockIdx.x * 128;

  // stage flat rows (128 x 128 fp32 -> bf16)
#pragma unroll
  for (int i = 0; i < 16; ++i) {
    int q   = t + i * 256;  // 4096 float4
    int row = q >> 5;       // 32 float4 per row
    int c4  = (q & 31) << 2;
    float4 v = *(const float4*)&enc[(size_t)(row0 + row) * 128 + c4];
    bf16_t* d = &As[row * SA + c4];
    d[0] = (bf16_t)v.x; d[1] = (bf16_t)v.y; d[2] = (bf16_t)v.z; d[3] = (bf16_t)v.w;
  }

  float minv[8];
  int   mini[8];
#pragma unroll
  for (int r = 0; r < 8; ++r) { minv[r] = 3.0e38f; mini[r] = 0; }

  for (int nc = 0; nc < 8; ++nc) {  // 8 chunks of 64 codebook rows
    __syncthreads();
#pragma unroll
    for (int i = 0; i < 8; ++i) {   // 2048 float4
      int q   = t + i * 256;
      int row = q >> 5;
      int c4  = (q & 31) << 2;
      float4 v = *(const float4*)&cb[(size_t)(nc * 64 + row) * 128 + c4];
      bf16_t* d = &Bs[row * SA + c4];
      d[0] = (bf16_t)v.x; d[1] = (bf16_t)v.y; d[2] = (bf16_t)v.z; d[3] = (bf16_t)v.w;
    }
    __syncthreads();

    v8f acc[4];
#pragma unroll
    for (int j = 0; j < 4; ++j)
#pragma unroll
      for (int r = 0; r < 8; ++r) acc[j][r] = 0.0f;

#pragma unroll
    for (int ks = 0; ks < 4; ++ks) {
      v16bf a = load_fragA(&As[(w * 16) * SA + ks * 32], SA);
#pragma unroll
      for (int j = 0; j < 4; ++j) {
        v16bf b = load_fragB(&Bs[(j * 16) * SA + ks * 32], SA);
        acc[j] = WMMA_BF16(a, b, acc[j]);
      }
    }

#pragma unroll
    for (int j = 0; j < 4; ++j) {
      int cn = nc * 64 + j * 16 + (lane & 15);
      float e = esq[cn];
#pragma unroll
      for (int r = 0; r < 8; ++r) {
        float d = e - 2.0f * acc[j][r];
        if (d < minv[r]) { minv[r] = d; mini[r] = cn; }
      }
    }
  }

  // argmin across the 16 lanes sharing each row (bits 0..3 of lane)
#pragma unroll
  for (int r = 0; r < 8; ++r) {
#pragma unroll
    for (int m = 1; m < 16; m <<= 1) {
      float ov = __shfl_xor(minv[r], m, 32);
      int   oi = __shfl_xor(mini[r], m, 32);
      if (ov < minv[r] || (ov == minv[r] && oi < mini[r])) { minv[r] = ov; mini[r] = oi; }
    }
  }
  if ((lane & 15) == 0) {
    int rbase = w * 16 + (lane >> 4) * 8;
#pragma unroll
    for (int r = 0; r < 8; ++r) {
      int lrow = rbase + r;
      int grow = row0 + lrow;
      idx_sh[lrow]   = mini[r];
      idx_out[grow]  = mini[r];
      idx_outf[grow] = (float)mini[r];
      atomicAdd(&counts[mini[r]], 1.0f);
    }
  }
  __syncthreads();

  // MSE partial: sum (cb[idx][c] - flat[c])^2 over the block's 128x128 elems
  float local = 0.0f;
#pragma unroll
  for (int i = 0; i < 64; ++i) {
    int e   = t + i * 256;
    int row = e >> 7;
    int c   = e & 127;
    int id  = idx_sh[row];
    float d = cb[(size_t)id * 128 + c] - enc[(size_t)(row0 + row) * 128 + c];
    local += d * d;
  }
#pragma unroll
  for (int m = 1; m < 32; m <<= 1) local += __shfl_xor(local, m, 32);
  if (lane == 0) atomicAdd(mse_acc, local);
}

// ---------------------------------------------------------------------------
__global__ void init_kernel(float* counts, float* mse_acc) {
  int t = blockIdx.x * blockDim.x + threadIdx.x;
  if (t < 512) counts[t] = 0.0f;
  if (t == 0) *mse_acc = 0.0f;
}

__global__ void esq_kernel(const float* __restrict__ cb, float* __restrict__ esq) {
  int t = blockIdx.x * blockDim.x + threadIdx.x;
  if (t < 512) {
    float s = 0.0f;
    for (int c = 0; c < 128; ++c) { float v = cb[t * 128 + c]; s += v * v; }
    esq[t] = s;
  }
}

__global__ void __launch_bounds__(512)
finalize_kernel(const float* __restrict__ counts, const float* __restrict__ mse_acc,
                float* __restrict__ out5) {
  __shared__ float sent[16], sused[16];
  const int t = threadIdx.x, lane = t & 31, w = t >> 5;
  float cnt  = counts[t];
  float avg  = cnt * (1.0f / 131072.0f);
  float ent  = avg * logf(avg + 1e-10f);
  float used = (cnt > 0.0f) ? 1.0f : 0.0f;
#pragma unroll
  for (int m = 1; m < 32; m <<= 1) { ent += __shfl_xor(ent, m, 32); used += __shfl_xor(used, m, 32); }
  if (lane == 0) { sent[w] = ent; sused[w] = used; }
  __syncthreads();
  if (t == 0) {
    float se = 0.0f, su = 0.0f;
    for (int i = 0; i < 16; ++i) { se += sent[i]; su += sused[i]; }
    float mse = *mse_acc * (1.0f / 16777216.0f);  // mean over N*C = 131072*128
    out5[0] = 1.25f * mse;   // loss = codebook + 0.25*commitment (equal values)
    out5[1] = mse;           // commitment_loss
    out5[2] = mse;           // codebook_loss
    out5[3] = expf(-se);     // perplexity
    out5[4] = su * (1.0f / 512.0f);  // usage_rate
  }
}

// ---------------------------------------------------------------------------
extern "C" void kernel_launch(void* const* d_in, const int* in_sizes, int n_in,
                              void* d_out, int out_size, void* d_ws, size_t ws_size,
                              hipStream_t stream) {
  (void)in_sizes; (void)n_in; (void)out_size; (void)ws_size;
  const float* h     = (const float*)d_in[0];
  const float* enc_w = (const float*)d_in[1];
  const float* enc_b = (const float*)d_in[2];
  const float* cb    = (const float*)d_in[3];
  const float* dec_w = (const float*)d_in[4];
  const float* dec_b = (const float*)d_in[5];
  float* out = (float*)d_out;

  const int M = 32768;   // B*T
  const int KC = 512;    // MSG_LEN*CODEBOOK_DIM
  const int D = 2048;
  const int NROWS = 131072;  // M*4 flat rows

  char* ws = (char*)d_ws;
  float* enc    = (float*)ws;                                      // M*KC fp32 (64MB)
  int*   idx    = (int*)(ws + (size_t)M * KC * 4);                 // NROWS
  float* esq    = (float*)(ws + (size_t)M * KC * 4 + (size_t)NROWS * 4);
  float* counts = esq + 512;
  float* mse    = counts + 512;

  float* message = out;                        // [M * D]
  float* idxf    = out + (size_t)M * D;        // [NROWS]
  float* out5    = idxf + NROWS;               // 5 scalars

  init_kernel<<<2, 256, 0, stream>>>(counts, mse);
  esq_kernel<<<2, 256, 0, stream>>>(cb, esq);

  // encoder: enc = h @ enc_w^T + enc_b
  gemm_bf16_kernel<false><<<dim3(KC / 128, M / 128), 256, 0, stream>>>(
      h, nullptr, nullptr, enc_w, enc_b, enc, M, KC, D);

  // quantize
  vq_kernel<<<NROWS / 128, 256, 0, stream>>>(enc, cb, esq, idx, idxf, counts, mse);

  // decoder: message = gather(cb, idx) @ dec_w^T + dec_b
  gemm_bf16_kernel<true><<<dim3(D / 128, M / 128), 256, 0, stream>>>(
      nullptr, idx, cb, dec_w, dec_b, message, M, D, KC);

  finalize_kernel<<<1, 512, 0, stream>>>(counts, mse, out5);
}
